// Attention_7876970021303
// MI455X (gfx1250) — matrix operand order repaired and checked by reading, not compiled, not run on hardware
//
#include <hip/hip_runtime.h>
#include <hip/hip_bf16.h>

typedef __attribute__((ext_vector_type(16))) _Float16 v16h;
typedef __attribute__((ext_vector_type(8)))  _Float16 v8h;
typedef __attribute__((ext_vector_type(4)))  _Float16 v4h;
typedef __attribute__((ext_vector_type(8)))  float    v8f;
typedef __attribute__((ext_vector_type(4)))  float    v4f;

#define B_  4
#define S_  2048
#define D_  1024
#define H_  16
#define DH_ 64
#define M_  (B_ * S_)   // 8192 rows for the big GEMMs

// ---- gfx1250 async global->LDS path (guarded; falls back to direct loads) --
#if defined(__gfx1250__) && __has_builtin(__builtin_amdgcn_global_load_async_to_lds_b128)
#define HAVE_ASYNC_LDS 1
#endif

#if defined(HAVE_ASYNC_LDS)
// Builtin signature (from hipcc diagnostic): (int4 AS1*, int4 AS3*, imm, imm)
typedef int v4i __attribute__((vector_size(16)));
typedef __attribute__((address_space(1))) v4i gv4i;  // global int4
typedef __attribute__((address_space(3))) v4i lv4i;  // LDS int4
#if __has_builtin(__builtin_amdgcn_s_wait_asynccnt)
#define WAIT_ASYNCCNT(n) __builtin_amdgcn_s_wait_asynccnt(n)
#else
#define WAIT_ASYNCCNT(n) asm volatile("s_wait_asynccnt %0" ::"n"(n) : "memory")
#endif
#endif

union AFrag { v16h v; v8h h[2]; };

// A-fragment (16x32 f16, MxK): lane L holds row M=(L&15); halves 0..7 are
// K = 8*(L>>4) .. +7, halves 8..15 are K = 8*(L>>4)+16 .. +23 (ISA 7.12.2).
__device__ __forceinline__ v16h load_a_frag(const _Float16* __restrict__ base,
                                            int row, int ld, int kBase, int lane) {
  const _Float16* p = base + (size_t)row * ld + kBase + ((lane >> 4) << 3);
  AFrag f;
  f.h[0] = *(const v8h*)p;
  f.h[1] = *(const v8h*)(p + 16);
  return f.v;
}

// B-fragment (32x16 f16, KxN): lane L holds column N=(L&15); halves 0..7 are
// K = 8*(L>>4) .. +7, halves 8..15 are K = 8*(L>>4)+16 .. +23. B "columns" are
// stored as contiguous rows [N][K], so two 16-byte loads per lane.
__device__ __forceinline__ v16h load_b_frag(const _Float16* __restrict__ base,
                                            int col, int ld, int kBase, int lane) {
  const _Float16* p = base + (size_t)col * ld + kBase + ((lane >> 4) << 3);
  AFrag f;
  f.h[0] = *(const v8h*)p;
  f.h[1] = *(const v8h*)(p + 16);
  return f.v;
}

#if defined(HAVE_ASYNC_LDS)
// Async-copy a contiguous 2048-byte block (one 16-key x 64-half K tile) from
// global memory to LDS: 4 x b128 per lane, tracked with ASYNCcnt.
__device__ __forceinline__ void async_copy_ktile(const _Float16* g, _Float16* l,
                                                 int lane) {
#pragma unroll
  for (int c = 0; c < 4; ++c) {
    const _Float16* gp = g + (c * 32 + lane) * 8;  // 8 halves = 16 B
    _Float16*       lp = l + (c * 32 + lane) * 8;
    __builtin_amdgcn_global_load_async_to_lds_b128((gv4i*)gp, (lv4i*)lp, 0, 0);
  }
}
#endif

// ---------------------------------------------------------------------------
// Generic WMMA GEMM: C[M,N] = A[M,K] (f16, row-major) * W[N,K]^T (f16).
// One wave computes a 16x64 C tile (A fragment reused across 4 N tiles).
// MODE 0: f32 plain [M,D] output (final projection -> d_out)
// MODE 1: f16 output scattered to [B,H,S,DH]   (Q, K)
// MODE 2: f16 output scattered to [B,H,DH,S]   (V transposed)
// ---------------------------------------------------------------------------
enum { OUT_F32_PLAIN = 0, OUT_F16_BHSD = 1, OUT_F16_BHDS = 2 };

template <int MODE>
__global__ __launch_bounds__(32)
void gemm_wmma(const _Float16* __restrict__ A, const _Float16* __restrict__ W,
               void* __restrict__ outp, int K) {
  const int lane  = threadIdx.x;
  const int mBase = blockIdx.x * 16;
  const int nQuad = blockIdx.y * 64;
  const int row   = mBase + (lane & 15);

  v8f acc[4] = {};
  for (int kt = 0; kt < K; kt += 32) {
    if (kt + 32 < K)  // gfx1250 global_prefetch of next A slice
      __builtin_prefetch(A + (size_t)row * K + kt + 32, 0, 0);
    v16h a = load_a_frag(A, row, K, kt, lane);
#pragma unroll
    for (int q = 0; q < 4; ++q) {
      v16h b = load_b_frag(W, nQuad + q * 16 + (lane & 15), K, kt, lane);
      acc[q] = __builtin_amdgcn_wmma_f32_16x16x32_f16(
          false, a, false, b, (short)0, acc[q], false, false);
    }
  }

#pragma unroll
  for (int q = 0; q < 4; ++q) {
#pragma unroll
    for (int r = 0; r < 8; ++r) {
      const int m = mBase + r + ((lane >> 4) << 3);
      const int n = nQuad + q * 16 + (lane & 15);
      const float v = acc[q][r];
      if (MODE == OUT_F32_PLAIN) {
        ((float*)outp)[(size_t)m * D_ + n] = v;
      } else {
        const int b  = m / S_, s = m % S_;
        const int h  = n / DH_, dh = n % DH_;
        const size_t off = (MODE == OUT_F16_BHSD)
            ? (((size_t)(b * H_ + h) * S_ + s) * DH_ + dh)
            : (((size_t)(b * H_ + h) * DH_ + dh) * S_ + s);
        ((_Float16*)outp)[off] = (_Float16)v;
      }
    }
  }
}

// ---------------------------------------------------------------------------
// RoPE applied in-place to Q and K ([B,H,S,DH] f16), math in f32.
// ---------------------------------------------------------------------------
__global__ __launch_bounds__(256)
void rope_qk(_Float16* __restrict__ Q, _Float16* __restrict__ Kk,
             const float* __restrict__ cosT, const float* __restrict__ sinT) {
  const int t = blockIdx.x * blockDim.x + threadIdx.x;
  if (t >= B_ * H_ * S_ * (DH_ / 2)) return;
  const int i  = t & 31;              // rotation-pair index
  const int s  = (t >> 5) & (S_ - 1); // sequence position
  const int bh = t >> 16;             // fused (b*H + h)
  const size_t base = ((size_t)bh * S_ + s) * DH_;
  const float c1 = cosT[s * DH_ + i],      s1 = sinT[s * DH_ + i];
  const float c2 = cosT[s * DH_ + i + 32], s2 = sinT[s * DH_ + i + 32];
  {
    const float a = (float)Q[base + i], b = (float)Q[base + i + 32];
    Q[base + i]      = (_Float16)(a * c1 - b * s1);
    Q[base + i + 32] = (_Float16)(b * c2 + a * s2);
  }
  {
    const float a = (float)Kk[base + i], b = (float)Kk[base + i + 32];
    Kk[base + i]      = (_Float16)(a * c1 - b * s1);
    Kk[base + i + 32] = (_Float16)(b * c2 + a * s2);
  }
}

// ---------------------------------------------------------------------------
// Attention: one wave per (b, h, 16-query tile).
//  Pass 1: K tiles double-buffered into LDS with GLOBAL_LOAD_ASYNC_TO_LDS
//          (ASYNCcnt-synchronized) feeding WMMA QK^T into an f32 LDS strip.
//  Softmax: 2 lanes per row, 4-wide vectorized; probs stored UNNORMALIZED in
//           f16 (exp(s-max) in [0,1]); 1/rowsum folded into the PV epilogue.
//  Pass 2: out = P (16 x S, f16 in LDS) x V^T-stored (DH x S) via WMMA.
// Dynamic LDS: 256 B red + 128 KB scores + 64 KB probs + 4 KB kbuf = 200960 B.
// ---------------------------------------------------------------------------
__global__ __launch_bounds__(32)
void attn_wmma(const _Float16* __restrict__ Q, const _Float16* __restrict__ Kc,
               const _Float16* __restrict__ Vt, _Float16* __restrict__ O) {
  extern __shared__ char smem[];
  float*    red = (float*)smem;                              // 64 floats
  float*    sc  = (float*)(smem + 256);                      // [16][S_] f32
  _Float16* pr  = (_Float16*)(smem + 256 + 16 * S_ * 4);     // [16][S_] f16

  const int lane = threadIdx.x;
  const int idx  = blockIdx.x;
  const int qt   = idx & 127;         // S_/16 = 128 query tiles
  const int h    = (idx >> 7) & 15;
  const int b    = idx >> 11;
  const int qBase = qt * 16;

  const _Float16* Qb = Q  + ((size_t)(b * H_ + h)) * S_ * DH_;
  const _Float16* Kb = Kc + ((size_t)(b * H_ + h)) * S_ * DH_;
  const _Float16* Vb = Vt + ((size_t)(b * H_ + h)) * DH_ * S_;

  // ---- Pass 1: scores = Q K^T / sqrt(DH) --------------------------------
  const int mRow = qBase + (lane & 15);
#if defined(HAVE_ASYNC_LDS)
  _Float16* kbuf = pr + 16 * S_;  // 2 x [16][DH_] halves (double buffer)
  async_copy_ktile(Kb, kbuf, lane);  // prefetch tile 0
#endif
  for (int kt = 0; kt < S_ / 16; ++kt) {
#if defined(HAVE_ASYNC_LDS)
    const _Float16* ktile = kbuf + (kt & 1) * (16 * DH_);
    if (kt + 1 < S_ / 16) {
      async_copy_ktile(Kb + (size_t)(kt + 1) * 16 * DH_,
                       kbuf + ((kt + 1) & 1) * (16 * DH_), lane);
      WAIT_ASYNCCNT(4);  // in-order: current tile's 4 transfers are complete
    } else {
      WAIT_ASYNCCNT(0);
    }
    asm volatile("" ::: "memory");  // keep LDS reads below the wait
#endif
    v8f c = {};
#pragma unroll
    for (int kk = 0; kk < DH_; kk += 32) {
      v16h a = load_a_frag(Qb, mRow, DH_, kk, lane);  // loop-invariant: hoisted
#if defined(HAVE_ASYNC_LDS)
      v16h bm = load_b_frag(ktile, lane & 15, DH_, kk, lane);
#else
      v16h bm = load_b_frag(Kb, kt * 16 + (lane & 15), DH_, kk, lane);
#endif
      c = __builtin_amdgcn_wmma_f32_16x16x32_f16(
          false, a, false, bm, (short)0, c, false, false);
    }
#pragma unroll
    for (int r = 0; r < 8; ++r)
      sc[(r + ((lane >> 4) << 3)) * S_ + kt * 16 + (lane & 15)] = c[r] * 0.125f;
  }
  __syncthreads();

  // ---- Softmax stats (2 lanes per row, float4 vectorized) ---------------
  {
    const int row = lane & 15, half = lane >> 4;
    const v4f* s4 = (const v4f*)(sc + row * S_ + half * (S_ / 2));
    v4h*       p4 = (v4h*)(pr + row * S_ + half * (S_ / 2));

    // row max
    float mx = -3.4e38f;
    for (int j = 0; j < (S_ / 2) / 4; ++j) {
      const v4f v = s4[j];
      mx = fmaxf(mx, fmaxf(fmaxf(v[0], v[1]), fmaxf(v[2], v[3])));
    }
    red[half * 16 + row] = mx;
    __syncthreads();
    const float mrow = fmaxf(red[row], red[16 + row]);

    // exp + partial sum; store unnormalized probs as f16
    v4f acc4 = {};
    for (int j = 0; j < (S_ / 2) / 4; ++j) {
      const v4f v = s4[j];
      v4f e;
      e[0] = __expf(v[0] - mrow);
      e[1] = __expf(v[1] - mrow);
      e[2] = __expf(v[2] - mrow);
      e[3] = __expf(v[3] - mrow);
      acc4 += e;
      v4h hh;
      hh[0] = (_Float16)e[0];
      hh[1] = (_Float16)e[1];
      hh[2] = (_Float16)e[2];
      hh[3] = (_Float16)e[3];
      p4[j] = hh;
    }
    red[32 + half * 16 + row] = acc4[0] + acc4[1] + acc4[2] + acc4[3];
    __syncthreads();
    if (half == 0)  // combined reciprocal row-sum -> red[0..15]
      red[row] = 1.0f / (red[32 + row] + red[48 + row]);
  }
  __syncthreads();

  // ---- Pass 2: out = P x V  (V stored transposed: contiguous B frags) ---
  v8f acc[4] = {};
  for (int kt = 0; kt < S_; kt += 32) {
    v16h a;
    {
      const _Float16* p = pr + (size_t)(lane & 15) * S_ + kt + ((lane >> 4) << 3);
      AFrag f;
      f.h[0] = *(const v8h*)p;
      f.h[1] = *(const v8h*)(p + 16);
      a = f.v;
    }
#pragma unroll
    for (int q = 0; q < 4; ++q) {
      v16h bm = load_b_frag(Vb, q * 16 + (lane & 15), S_, kt, lane);
      acc[q] = __builtin_amdgcn_wmma_f32_16x16x32_f16(
          false, a, false, bm, (short)0, acc[q], false, false);
    }
  }
#pragma unroll
  for (int q = 0; q < 4; ++q) {
#pragma unroll
    for (int r = 0; r < 8; ++r) {
      const int rloc = r + ((lane >> 4) << 3);
      const int m    = qBase + rloc;
      const int dh   = q * 16 + (lane & 15);
      const float inv = red[rloc];  // 1/rowsum folded in here
      O[((size_t)b * S_ + m) * D_ + h * DH_ + dh] = (_Float16)(acc[q][r] * inv);
    }
  }
}

// ---------------------------------------------------------------------------
__global__ __launch_bounds__(256)
void cvt_f32_f16(const float* __restrict__ in, _Float16* __restrict__ out, int n4) {
  const int i = blockIdx.x * blockDim.x + threadIdx.x;  // one float4 per thread
  if (i < n4) {
    const v4f v = ((const v4f*)in)[i];
    v4h hh;
    hh[0] = (_Float16)v[0];
    hh[1] = (_Float16)v[1];
    hh[2] = (_Float16)v[2];
    hh[3] = (_Float16)v[3];
    ((v4h*)out)[i] = hh;
  }
}

extern "C" void kernel_launch(void* const* d_in, const int* in_sizes, int n_in,
                              void* d_out, int out_size, void* d_ws, size_t ws_size,
                              hipStream_t stream) {
  const float* x    = (const float*)d_in[0];
  const float* cosT = (const float*)d_in[1];
  const float* sinT = (const float*)d_in[2];
  const float* wq   = (const float*)d_in[3];
  const float* wk   = (const float*)d_in[4];
  const float* wv   = (const float*)d_in[5];
  const float* wo   = (const float*)d_in[6];
  float* out = (float*)d_out;

  // Workspace carve-up (~92 MB)
  char* ws = (char*)d_ws;
  _Float16* xh  = (_Float16*)ws; ws += (size_t)M_ * D_ * 2;
  _Float16* wqh = (_Float16*)ws; ws += (size_t)D_ * D_ * 2;
  _Float16* wkh = (_Float16*)ws; ws += (size_t)D_ * D_ * 2;
  _Float16* wvh = (_Float16*)ws; ws += (size_t)D_ * D_ * 2;
  _Float16* woh = (_Float16*)ws; ws += (size_t)D_ * D_ * 2;
  _Float16* Qh  = (_Float16*)ws; ws += (size_t)M_ * D_ * 2;  // [B,H,S,DH]
  _Float16* Kh  = (_Float16*)ws; ws += (size_t)M_ * D_ * 2;  // [B,H,S,DH]
  _Float16* Vth = (_Float16*)ws; ws += (size_t)M_ * D_ * 2;  // [B,H,DH,S]
  _Float16* Ah  = (_Float16*)ws; ws += (size_t)M_ * D_ * 2;  // [B,S,D]

  // 1) f32 -> f16 conversions (float4-wide)
  cvt_f32_f16<<<(M_ * D_ / 4) / 256, 256, 0, stream>>>(x, xh, M_ * D_ / 4);
  cvt_f32_f16<<<(D_ * D_ / 4) / 256, 256, 0, stream>>>(wq, wqh, D_ * D_ / 4);
  cvt_f32_f16<<<(D_ * D_ / 4) / 256, 256, 0, stream>>>(wk, wkh, D_ * D_ / 4);
  cvt_f32_f16<<<(D_ * D_ / 4) / 256, 256, 0, stream>>>(wv, wvh, D_ * D_ / 4);
  cvt_f32_f16<<<(D_ * D_ / 4) / 256, 256, 0, stream>>>(wo, woh, D_ * D_ / 4);

  // 2) Q/K/V projections via WMMA
  dim3 gGemm(M_ / 16, D_ / 64);
  gemm_wmma<OUT_F16_BHSD><<<gGemm, 32, 0, stream>>>(xh, wqh, (void*)Qh, D_);
  gemm_wmma<OUT_F16_BHSD><<<gGemm, 32, 0, stream>>>(xh, wkh, (void*)Kh, D_);
  gemm_wmma<OUT_F16_BHDS><<<gGemm, 32, 0, stream>>>(xh, wvh, (void*)Vth, D_);

  // 3) RoPE on Q and K
  rope_qk<<<(B_ * H_ * S_ * (DH_ / 2)) / 256, 256, 0, stream>>>(Qh, Kh, cosT, sinT);

  // 4) Attention (dynamic LDS: 256 + 128K + 64K + 4K)
  const size_t shmem = 256 + (size_t)16 * S_ * 4 + (size_t)16 * S_ * 2 + 4096;
  (void)hipFuncSetAttribute((const void*)attn_wmma,
                            hipFuncAttributeMaxDynamicSharedMemorySize, (int)shmem);
  attn_wmma<<<B_ * H_ * (S_ / 16), 32, shmem, stream>>>(Qh, Kh, Vth, Ah);

  // 5) Output projection -> f32 d_out
  gemm_wmma<OUT_F32_PLAIN><<<gGemm, 32, 0, stream>>>(Ah, woh, (void*)out, D_);

  (void)in_sizes; (void)n_in; (void)out_size; (void)ws_size;
}
